// TransformerXLEncoderLayer_73710228734052
// MI455X (gfx1250) — compile-verified, matrix operand order checked by reading, not run on hardware
//
#include <hip/hip_runtime.h>

#define DIMN 2048
#define HEADS 16
#define HDIM 128
#define FFN 8192
#define MAX_SEQ 4096
#define BB 2
#define SS 2048
#define PREVN 1024
#define TLEN (PREVN + SS)      // 3072
#define BSR (BB * SS)          // 4096 rows
#define EPSF 1e-6f

typedef __attribute__((ext_vector_type(16))) _Float16 v16h;
typedef __attribute__((ext_vector_type(8)))  float    v8f;
typedef __attribute__((ext_vector_type(4)))  unsigned int uint32x4;
typedef __attribute__((ext_vector_type(8)))  int      int32x8;
typedef __attribute__((ext_vector_type(4)))  int      int32x4;

#if defined(__has_builtin)
#  if __has_builtin(__builtin_amdgcn_tensor_load_to_lds) && \
      __has_builtin(__builtin_amdgcn_s_wait_tensorcnt)
#    define HAVE_TDM 1
#  endif
#endif
#ifndef HAVE_TDM
#  define HAVE_TDM 0
#endif

#if HAVE_TDM
// Issue one TDM 2-D tile load: global (row-major, elem=2B, row stride d0_stride
// elems) -> contiguous LDS at lds_off. D# packing per CDNA5 ISA 08_async_tensor §8.
__device__ __forceinline__ void tdm_load_2d(const void* gptr, unsigned int lds_off,
                                            unsigned int tensor_d0, unsigned int tensor_d1,
                                            unsigned int tile_d0, unsigned int tile_d1,
                                            unsigned long long d0_stride) {
  unsigned long long ga = (unsigned long long)(uintptr_t)gptr;
  uint32x4 g0;
  g0[0] = 1u;                                            // count=1, user descriptor
  g0[1] = lds_off;                                       // lds_addr (bytes)
  g0[2] = (unsigned int)(ga & 0xffffffffu);              // global_addr[31:0]
  g0[3] = (unsigned int)((ga >> 32) & 0x01ffffffu)       // global_addr[56:32]
        | (2u << 30);                                    // type = 2 ("image")
  int32x8 g1;
  g1[0] = (int)(1u << 16);                               // wg_mask=0, data_size=1 (2B)
  g1[1] = (int)((tensor_d0 & 0xffffu) << 16);            // tensor_dim0[15:0]
  g1[2] = (int)(((tensor_d0 >> 16) & 0xffffu) |          // tensor_dim0[31:16]
                ((tensor_d1 & 0xffffu) << 16));          // tensor_dim1[15:0]
  g1[3] = (int)(((tensor_d1 >> 16) & 0xffffu) |          // tensor_dim1[31:16]
                ((tile_d0 & 0xffffu) << 16));            // tile_dim0
  g1[4] = (int)(tile_d1 & 0xffffu);                      // tile_dim1 (tile_dim2=0)
  g1[5] = (int)(unsigned int)(d0_stride & 0xffffffffu);  // dim0_stride[31:0]
  g1[6] = (int)(unsigned int)((d0_stride >> 32) & 0xffffu); // dim0_stride[47:32]
  g1[7] = 0;                                             // dim1_stride (unused, 2D)
  int32x4 z4 = {0, 0, 0, 0};                             // groups 2/3: <=2D tensor
#if __clang_major__ >= 23
  int32x8 z8 = {0, 0, 0, 0, 0, 0, 0, 0};
  __builtin_amdgcn_tensor_load_to_lds(g0, g1, z4, z4, z8, 0);
#else
  __builtin_amdgcn_tensor_load_to_lds(g0, g1, z4, z4, 0);
#endif
}
#endif

// Load a 16-half WMMA fragment from two contiguous 16-byte chunks.
// Element mapping (CDNA5 16-bit A/B layout): e in [0,7] -> K = hi*8+e,
// e in [8,15] -> K = 16 + hi*8 + (e-8). lo points at K-base, hi8 at K-base+16.
__device__ __forceinline__ v16h frag_ld(const _Float16* lo, const _Float16* hi8) {
  v16h r;
  ((uint4*)&r)[0] = *(const uint4*)lo;
  ((uint4*)&r)[1] = *(const uint4*)hi8;
  return r;
}

// ---------------- RMSNorm: f32 row -> f16 row --------------------------------
__global__ __launch_bounds__(256)
void rmsnorm_kernel(const float* __restrict__ x, const float* __restrict__ w,
                    _Float16* __restrict__ out) {
  int row = blockIdx.x;
  int tid = threadIdx.x;
  const float* xr = x + (size_t)row * DIMN;
  float4 a = ((const float4*)xr)[tid * 2];
  float4 b = ((const float4*)xr)[tid * 2 + 1];
  float ss = a.x*a.x + a.y*a.y + a.z*a.z + a.w*a.w
           + b.x*b.x + b.y*b.y + b.z*b.z + b.w*b.w;
  for (int m = 16; m >= 1; m >>= 1) ss += __shfl_xor(ss, m, 32);
  __shared__ float wsum[8];
  int lane = tid & 31, wid = tid >> 5;
  if (lane == 0) wsum[wid] = ss;
  __syncthreads();
  float tot = 0.f;
  #pragma unroll
  for (int i = 0; i < 8; i++) tot += wsum[i];
  float scale = rsqrtf(tot / (float)DIMN + EPSF);
  int c0 = tid * 8;
  const float* wr = w + c0;
  float xv[8] = {a.x,a.y,a.z,a.w,b.x,b.y,b.z,b.w};
  _Float16 h[8];
  #pragma unroll
  for (int j = 0; j < 8; j++) h[j] = (_Float16)(xv[j] * scale * wr[j]);
  *(uint4*)&out[(size_t)row * DIMN + c0] = *(const uint4*)h;
}

// ---------------- Generic f16 WMMA GEMM: C[M,N] = A[M,K] @ W[K,N] + bias -----
// mode 0: out f32      1: out f32 + residual     2: out f16
// mode 3: silu -> f16  4: (aux f16 * val) -> f16 (fused SwiGLU multiply)
__global__ __launch_bounds__(256)
void gemm_f16_kernel(const _Float16* __restrict__ A, const float* __restrict__ W,
                     const float* __restrict__ bias,
                     float* __restrict__ outF, const float* __restrict__ resid,
                     _Float16* __restrict__ outH, const _Float16* __restrict__ aux,
                     int M, int N, int K, int mode) {
  __shared__ __align__(16) _Float16 sA[128 * 32];     // [m][k]
  __shared__ __align__(16) _Float16 sW[64 * 32];      // transposed [n][k]
  int tid = threadIdx.x;
  int lane = tid & 31, wid = tid >> 5;
  int m0 = blockIdx.y * 128, n0 = blockIdx.x * 64;
  int rlo = lane & 15, hi = lane >> 4;
  v8f acc[4] = {};
  for (int k0 = 0; k0 < K; k0 += 32) {
#if HAVE_TDM
    // A tile via Tensor Data Mover: 128x32 f16 tile, row stride K elems.
    if (tid < 32) {
      tdm_load_2d(&A[(size_t)m0 * K + k0], (unsigned int)(uintptr_t)sA,
                  (unsigned int)K, 128u, 32u, 128u, (unsigned long long)K);
    }
#else
    // stage A tile: 128x32 halves, 8 halves (16B) per chunk, 2 chunks/thread
    #pragma unroll
    for (int i = 0; i < 2; i++) {
      int c = tid * 2 + i;
      int row = c >> 2, off = (c & 3) * 8;
      *(uint4*)&sA[row * 32 + off] =
          *(const uint4*)&A[(size_t)(m0 + row) * K + k0 + off];
    }
#endif
    // stage W tile transposed: 32(k) x 64(n) f32 -> sW[n][k] f16
    #pragma unroll
    for (int i = 0; i < 2; i++) {
      int c = tid * 2 + i;
      int kr = c >> 4, nc = (c & 15) * 4;
      float4 w4 = *(const float4*)&W[(size_t)(k0 + kr) * N + n0 + nc];
      sW[(nc + 0) * 32 + kr] = (_Float16)w4.x;
      sW[(nc + 1) * 32 + kr] = (_Float16)w4.y;
      sW[(nc + 2) * 32 + kr] = (_Float16)w4.z;
      sW[(nc + 3) * 32 + kr] = (_Float16)w4.w;
    }
    if (k0 + 32 < K) {   // prefetch next tiles (global_prefetch_b8 path)
      __builtin_prefetch(&A[(size_t)(m0 + (tid >> 1)) * K + k0 + 32], 0, 0);
      __builtin_prefetch(&W[(size_t)(k0 + 32 + (tid >> 4)) * N + n0 + (tid & 15) * 4], 0, 0);
    }
#if HAVE_TDM
    if (tid < 32) __builtin_amdgcn_s_wait_tensorcnt(0);
#endif
    __syncthreads();
    const _Float16* ap = &sA[(wid * 16 + rlo) * 32 + hi * 8];
    v16h af = frag_ld(ap, ap + 16);
    #pragma unroll
    for (int ns = 0; ns < 4; ns++) {
      const _Float16* bp = &sW[(ns * 16 + rlo) * 32 + hi * 8];
      v16h bf = frag_ld(bp, bp + 16);
      acc[ns] = __builtin_amdgcn_wmma_f32_16x16x32_f16(
          false, af, false, bf, (short)0, acc[ns], false, false);
    }
    __syncthreads();
  }
  #pragma unroll
  for (int ns = 0; ns < 4; ns++) {
    #pragma unroll
    for (int v = 0; v < 8; v++) {
      int gr = m0 + wid * 16 + v + 8 * hi;
      int gc = n0 + ns * 16 + rlo;
      size_t idx = (size_t)gr * N + gc;
      float val = acc[ns][v] + bias[gc];
      if (mode == 0)      outF[idx] = val;
      else if (mode == 1) outF[idx] = val + resid[idx];
      else if (mode == 2) outH[idx] = (_Float16)val;
      else if (mode == 3) outH[idx] = (_Float16)(val / (1.f + __expf(-val)));
      else                outH[idx] = (_Float16)((float)aux[idx] * val);
    }
  }
}

// ---------------- RoPE for Q: f32 [B,S,H*D] -> f16 [B,H,S,D] -----------------
__global__ __launch_bounds__(256)
void rope_q_kernel(const float* __restrict__ q, const float* __restrict__ fc,
                   const float* __restrict__ fs, const int* __restrict__ start_pos,
                   _Float16* __restrict__ qh) {
  size_t idx = (size_t)blockIdx.x * 256 + threadIdx.x;
  int d = idx & (HDIM - 1);
  size_t r = idx >> 7;
  int s = (int)(r % SS);
  size_t bh = r / SS;
  int h = (int)(bh % HEADS), b = (int)(bh / HEADS);
  size_t base = ((size_t)b * SS + s) * DIMN + h * HDIM;
  int p = d >> 1;
  int pos = (s + start_pos[0]) % MAX_SEQ;
  float c = fc[(size_t)pos * (HDIM / 2) + p];
  float sn = fs[(size_t)pos * (HDIM / 2) + p];
  float x0 = q[base + (d & ~1)], x1 = q[base + (d | 1)];
  float val = (d & 1) ? (x0 * sn + x1 * c) : (x0 * c - x1 * sn);
  qh[idx] = (_Float16)val;
}

// -------- Build k_attn/v_attn (f32 outputs) + f16 copies; RoPE new K ---------
__global__ __launch_bounds__(256)
void build_kv_kernel(const float* __restrict__ kcache, const float* __restrict__ vcache,
                     const float* __restrict__ knew, const float* __restrict__ vnew,
                     const float* __restrict__ fc, const float* __restrict__ fs,
                     const int* __restrict__ start_pos,
                     float* __restrict__ outK, float* __restrict__ outV,
                     _Float16* __restrict__ kh, _Float16* __restrict__ vh) {
  size_t idx = (size_t)blockIdx.x * 256 + threadIdx.x;
  int d = idx & (HDIM - 1);
  size_t r = idx >> 7;
  int t = (int)(r % TLEN);
  size_t bh = r / TLEN;
  float kval, vval;
  if (t < PREVN) {
    size_t ci = (bh * PREVN + t) * HDIM + d;
    kval = kcache[ci];
    vval = vcache[ci];
  } else {
    int s = t - PREVN;
    int h = (int)(bh % HEADS), b = (int)(bh / HEADS);
    size_t base = ((size_t)b * SS + s) * DIMN + h * HDIM;
    float x0 = knew[base + (d & ~1)], x1 = knew[base + (d | 1)];
    int pos = (s + start_pos[0]) % MAX_SEQ;
    int p = d >> 1;
    float c = fc[(size_t)pos * (HDIM / 2) + p];
    float sn = fs[(size_t)pos * (HDIM / 2) + p];
    kval = (d & 1) ? (x0 * sn + x1 * c) : (x0 * c - x1 * sn);
    vval = vnew[base + d];
  }
  outK[idx] = kval;
  outV[idx] = vval;
  kh[idx] = (_Float16)kval;
  vh[idx] = (_Float16)vval;
}

// ---------------- Flash attention (causal), all-WMMA -------------------------
// Block: 8 waves, one (b,h), 128 query rows; each wave owns 16 query rows.
__global__ __launch_bounds__(256)
void attn_kernel(const _Float16* __restrict__ qh, const _Float16* __restrict__ kh,
                 const _Float16* __restrict__ vh, _Float16* __restrict__ o16) {
  __shared__ __align__(16) _Float16 sK[32 * 128];  // [t][d]
  __shared__ __align__(16) _Float16 sV[128 * 32];  // transposed [d][t]
  __shared__ __align__(16) float sc[8][16 * 32];   // per-wave scores / probs
  __shared__ float sAlpha[8][16];
  __shared__ float sL[8][16];
  int tid = threadIdx.x, lane = tid & 31, wid = tid >> 5;
  int bh = blockIdx.x;
  int b = bh / HEADS, h = bh % HEADS;
  int qb = blockIdx.y * 128;
  int rlo = lane & 15, hi = lane >> 4;
  int srow = qb + wid * 16 + rlo;
  const _Float16* qrow = qh + ((size_t)bh * SS + srow) * HDIM;
  v16h aq[4];
  #pragma unroll
  for (int kc = 0; kc < 4; kc++)
    aq[kc] = frag_ld(qrow + kc * 32 + hi * 8, qrow + kc * 32 + 16 + hi * 8);
  v8f oacc[8] = {};
  float mrun = -3.0e38f, lrun = 0.f;  // row stats (lanes 0..15)
  const float rscale = 0.088388347648318447f;  // 1/sqrt(128)
  int ntiles = (PREVN + qb + 128) / 32;
  for (int tt = 0; tt < ntiles; tt++) {
    int t0 = tt * 32;
#if HAVE_TDM
    // K tile is a fully contiguous 8KB span -> 1-D TDM load into sK.
    if (tid < 32) {
      tdm_load_2d(&kh[((size_t)bh * TLEN + t0) * HDIM], (unsigned int)(uintptr_t)sK,
                  4096u, 1u, 4096u, 1u, 4096ull);
    }
#endif
    // stage V tile transposed [128][32] (and K manually if no TDM)
    #pragma unroll
    for (int i = 0; i < 2; i++) {
      int c = tid * 2 + i;
      int tr = c >> 4, off = (c & 15) * 8;
      size_t gbase = ((size_t)bh * TLEN + t0 + tr) * HDIM + off;
#if !HAVE_TDM
      *(uint4*)&sK[tr * 128 + off] = *(const uint4*)&kh[gbase];
#endif
      uint4 vv4 = *(const uint4*)&vh[gbase];
      const _Float16* vp = (const _Float16*)&vv4;
      #pragma unroll
      for (int j = 0; j < 8; j++) sV[(off + j) * 32 + tr] = vp[j];
    }
#if HAVE_TDM
    if (tid < 32) __builtin_amdgcn_s_wait_tensorcnt(0);
#endif
    __syncthreads();
    // scores: Q(16x128) @ K_tile^T(128x32) -> two 16x16 WMMA chains
    #pragma unroll
    for (int sub = 0; sub < 2; sub++) {
      v8f s = {};
      #pragma unroll
      for (int kc = 0; kc < 4; kc++) {
        const _Float16* bp = &sK[(sub * 16 + rlo) * 128 + kc * 32 + hi * 8];
        v16h bf = frag_ld(bp, bp + 16);
        s = __builtin_amdgcn_wmma_f32_16x16x32_f16(
            false, aq[kc], false, bf, (short)0, s, false, false);
      }
      #pragma unroll
      for (int v = 0; v < 8; v++)
        sc[wid][(v + 8 * hi) * 32 + sub * 16 + rlo] = s[v] * rscale;
    }
    __syncthreads();
    // online softmax: lanes 0..15 own one row each
    if (lane < 16) {
      int rr = lane;
      int limit = PREVN + qb + wid * 16 + rr;  // allowed: t <= limit
      float tmax = -3.0e38f;
      #pragma unroll
      for (int c = 0; c < 32; c++) {
        float v = (t0 + c > limit) ? -3.0e38f : sc[wid][rr * 32 + c];
        tmax = fmaxf(tmax, v);
      }
      float nm = fmaxf(mrun, tmax);
      float alpha = __expf(mrun - nm);
      float ps = 0.f;
      #pragma unroll
      for (int c = 0; c < 32; c++) {
        float v = sc[wid][rr * 32 + c];
        float p = (t0 + c > limit) ? 0.f : __expf(v - nm);
        sc[wid][rr * 32 + c] = p;
        ps += p;
      }
      lrun = lrun * alpha + ps;
      mrun = nm;
      sAlpha[wid][rr] = alpha;
    }
    __syncthreads();
    // rescale running accumulators
    #pragma unroll
    for (int v = 0; v < 8; v++) {
      float a = sAlpha[wid][v + 8 * hi];
      #pragma unroll
      for (int nc = 0; nc < 8; nc++) oacc[nc][v] *= a;
    }
    // P fragment (A-matrix 16x32) from probability scratch
    const float* pp = &sc[wid][rlo * 32 + hi * 8];
    v16h ap;
    #pragma unroll
    for (int e = 0; e < 8; e++) ap[e] = (_Float16)pp[e];
    #pragma unroll
    for (int e = 0; e < 8; e++) ap[8 + e] = (_Float16)pp[16 + e];
    // P(16x32) @ V_tile(32x128)
    #pragma unroll
    for (int nc = 0; nc < 8; nc++) {
      const _Float16* bp = &sV[(nc * 16 + rlo) * 32 + hi * 8];
      v16h bf = frag_ld(bp, bp + 16);
      oacc[nc] = __builtin_amdgcn_wmma_f32_16x16x32_f16(
          false, ap, false, bf, (short)0, oacc[nc], false, false);
    }
    __syncthreads();
  }
  if (lane < 16) sL[wid][lane] = lrun;
  __syncthreads();
  #pragma unroll
  for (int v = 0; v < 8; v++) {
    float inv = 1.f / sL[wid][v + 8 * hi];
    int s = qb + wid * 16 + v + 8 * hi;
    size_t base = ((size_t)b * SS + s) * DIMN + h * HDIM;
    #pragma unroll
    for (int nc = 0; nc < 8; nc++)
      o16[base + nc * 16 + rlo] = (_Float16)(oacc[nc][v] * inv);
  }
}

// ---------------- Host-side launch -------------------------------------------
extern "C" void kernel_launch(void* const* d_in, const int* in_sizes, int n_in,
                              void* d_out, int out_size, void* d_ws, size_t ws_size,
                              hipStream_t stream) {
  const float* x          = (const float*)d_in[0];
  const float* k_cache    = (const float*)d_in[1];
  const float* v_cache    = (const float*)d_in[2];
  const float* attn_nw    = (const float*)d_in[3];
  const float* ffn_nw     = (const float*)d_in[4];
  const float* wq = (const float*)d_in[5];   const float* bq = (const float*)d_in[6];
  const float* wk = (const float*)d_in[7];   const float* bk = (const float*)d_in[8];
  const float* wv = (const float*)d_in[9];   const float* bv = (const float*)d_in[10];
  const float* wo = (const float*)d_in[11];  const float* bo = (const float*)d_in[12];
  const float* w_gate = (const float*)d_in[13]; const float* b_gate = (const float*)d_in[14];
  const float* w_val  = (const float*)d_in[15]; const float* b_val  = (const float*)d_in[16];
  const float* w_proj = (const float*)d_in[17]; const float* b_proj = (const float*)d_in[18];
  const float* fcos = (const float*)d_in[19];
  const float* fsin = (const float*)d_in[20];
  const int* start_pos = (const int*)d_in[21];
  // d_in[22] = cache_limit (4096 >= TLEN, concat is a no-op truncation)

  char* ws = (char*)d_ws;
  const size_t OFF_H16 = 0;                 // 16 MB  f16 [BS,DIM]
  const size_t OFF_Q   = 16777216;          // 32 MB  f32 [BS,DIM]
  const size_t OFF_K   = 50331648;          // 32 MB  f32 [BS,DIM]
  const size_t OFF_V   = 83886080;          // 32 MB  f32 [BS,DIM]
  const size_t OFF_QH  = 117440512;         // 16 MB  f16 [B,H,S,D]
  const size_t OFF_KH  = 134217728;         // 24 MB  f16 [B,H,T,D]
  const size_t OFF_VH  = 159383552;         // 24 MB  f16 [B,H,T,D]
  const size_t OFF_FF  = 184549376;         // 64 MB  f16 [BS,FF]

  _Float16* h16  = (_Float16*)(ws + OFF_H16);
  float*    qf   = (float*)   (ws + OFF_Q);
  float*    kf   = (float*)   (ws + OFF_K);
  float*    vf   = (float*)   (ws + OFF_V);
  _Float16* qh16 = (_Float16*)(ws + OFF_QH);
  _Float16* kh16 = (_Float16*)(ws + OFF_KH);
  _Float16* vh16 = (_Float16*)(ws + OFF_VH);
  _Float16* ff16 = (_Float16*)(ws + OFF_FF);
  _Float16* o16  = (_Float16*)(ws + OFF_H16);  // reuse (h16 dead after QKV)
  float*    x1   = (float*)   (ws + OFF_Q);    // reuse (qf dead after rope_q)
  _Float16* h2   = (_Float16*)(ws + OFF_QH);   // reuse (qh16 dead after attn)
  _Float16* g16  = (_Float16*)(ws + OFF_K);    // reuse (kf+vf dead, 64MB span)

  float* outX = (float*)d_out;
  float* outK = outX + (size_t)BSR * DIMN;
  float* outV = outK + (size_t)BB * HEADS * TLEN * HDIM;

  dim3 blk(256);
  dim3 gdim(DIMN / 64, BSR / 128);
  dim3 gff(FFN / 64, BSR / 128);

  rmsnorm_kernel<<<BSR, blk, 0, stream>>>(x, attn_nw, h16);
  gemm_f16_kernel<<<gdim, blk, 0, stream>>>(h16, wq, bq, qf, nullptr, nullptr, nullptr,
                                            BSR, DIMN, DIMN, 0);
  gemm_f16_kernel<<<gdim, blk, 0, stream>>>(h16, wk, bk, kf, nullptr, nullptr, nullptr,
                                            BSR, DIMN, DIMN, 0);
  gemm_f16_kernel<<<gdim, blk, 0, stream>>>(h16, wv, bv, vf, nullptr, nullptr, nullptr,
                                            BSR, DIMN, DIMN, 0);
  rope_q_kernel<<<(BB * HEADS * SS * HDIM) / 256, blk, 0, stream>>>(qf, fcos, fsin,
                                                                    start_pos, qh16);
  build_kv_kernel<<<(BB * HEADS * TLEN * HDIM) / 256, blk, 0, stream>>>(
      k_cache, v_cache, kf, vf, fcos, fsin, start_pos, outK, outV, kh16, vh16);
  attn_kernel<<<dim3(BB * HEADS, SS / 128), blk, 0, stream>>>(qh16, kh16, vh16, o16);
  gemm_f16_kernel<<<gdim, blk, 0, stream>>>(o16, wo, bo, x1, x, nullptr, nullptr,
                                            BSR, DIMN, DIMN, 1);
  rmsnorm_kernel<<<BSR, blk, 0, stream>>>(x1, ffn_nw, h2);
  gemm_f16_kernel<<<gff, blk, 0, stream>>>(h2, w_gate, b_gate, nullptr, nullptr, g16,
                                           nullptr, BSR, FFN, DIMN, 3);
  gemm_f16_kernel<<<gff, blk, 0, stream>>>(h2, w_val, b_val, nullptr, nullptr, ff16,
                                           g16, BSR, FFN, DIMN, 4);
  gemm_f16_kernel<<<gdim, blk, 0, stream>>>(ff16, w_proj, b_proj, outX, x1, nullptr,
                                            nullptr, BSR, DIMN, FFN, 1);
}
